// ViTModelEcg_50070728737017
// MI455X (gfx1250) — matrix operand order repaired and checked by reading, not compiled, not run on hardware
//
#include <hip/hip_runtime.h>

// ---------------------------------------------------------------------------
// Types for CDNA5 WMMA (wave32, v_wmma_f32_16x16x32_bf16)
// ---------------------------------------------------------------------------
typedef __bf16        bf16x16 __attribute__((ext_vector_type(16)));
typedef float         f32x8   __attribute__((ext_vector_type(8)));
typedef float         f32x4   __attribute__((ext_vector_type(4)));
typedef unsigned int  u32x4   __attribute__((ext_vector_type(4)));
typedef int           v4i     __attribute__((vector_size(16)));   // for async-LDS builtin

union Frag { u32x4 u[2]; bf16x16 f; };

__device__ __forceinline__ f32x8 wmma_bf16(bf16x16 a, bf16x16 b, f32x8 c) {
    return __builtin_amdgcn_wmma_f32_16x16x32_bf16(false, a, false, b, (short)0, c, false, false);
}

__device__ __forceinline__ unsigned short f2bf(float x) {
    unsigned u = __builtin_bit_cast(unsigned, x);
    u += 0x7FFFu + ((u >> 16) & 1u);           // round-to-nearest-even
    return (unsigned short)(u >> 16);
}

// ---------------------------------------------------------------------------
// CDNA5 async global->LDS copy path (ASYNCcnt-tracked), guarded so the file
// still compiles if the toolchain lacks the builtins.
// Probe-derived signature: (v4i addrspace(1)*, v4i addrspace(3)*, imm, imm)
// ---------------------------------------------------------------------------
#if defined(__has_builtin)
#if __has_builtin(__builtin_amdgcn_global_load_async_to_lds_b128) && \
    __has_builtin(__builtin_amdgcn_s_wait_asynccnt)
#define USE_ASYNC_LDS 1
#endif
#endif
#ifndef USE_ASYNC_LDS
#define USE_ASYNC_LDS 0
#endif

#if USE_ASYNC_LDS
typedef __attribute__((address_space(1))) v4i* gv4i_t;
typedef __attribute__((address_space(3))) v4i* lv4i_t;
__device__ __forceinline__ void async_copy_b128(const void* g, void* l) {
    __builtin_amdgcn_global_load_async_to_lds_b128((gv4i_t)g, (lv4i_t)l, 0, 0);
}
__device__ __forceinline__ void async_wait0() {
    __builtin_amdgcn_s_wait_asynccnt(0);
}
#else
__device__ __forceinline__ void async_wait0() {}
#endif

// Model constants
#define NBATCH 64
#define DMODEL 768
#define NHEAD  12
#define HD     64
#define NTOK   197
#define MROWS  (NBATCH * NTOK)   // 12608

// Epilogue modes
#define EPI_BF16     0   // store bf16, no bias             (qkv)
#define EPI_BIAS_RES 1   // Cf[row,col] += acc + bias       (proj, fc2 into tok)
#define EPI_GELU     2   // bias + exact gelu -> bf16       (fc1)
#define EPI_PATCH    3   // bias,relu,BN,+pos -> tok scatter(conv2)
#define EPI_BN_RELU  4   // BN + relu -> bf16               (head p1)
#define EPI_BN_OUT   5   // BN -> f32                       (head p2)

// ---------------------------------------------------------------------------
// Generic bf16 WMMA GEMM: C[M,N] = A[M,K](bf16) * B[K,N](f32->bf16)
// Block: 256 thr = 8 waves (4x2), block tile 128x64, wave tile 32x32, K step 32
// Requires K%32==0, N%64==0. TRANSB: B stored [N][K] (ldb = K stride).
// ---------------------------------------------------------------------------
template <int EPI, bool TRANSB>
__launch_bounds__(256)
__global__ void gemm_kernel(const unsigned short* __restrict__ A,
                            const float* __restrict__ B,
                            float* __restrict__ Cf, unsigned short* __restrict__ Cb,
                            const float* __restrict__ bias,
                            const float* __restrict__ bng, const float* __restrict__ bnb,
                            const float* __restrict__ bnm, const float* __restrict__ bnv,
                            const float* __restrict__ pos,
                            int M, int N, int K, int lda, int ldb, int ldc) {
    __shared__ __align__(16) unsigned short Alds[128 * 40];
    __shared__ __align__(16) unsigned short Blds[64 * 40];

    const int tid  = threadIdx.x;
    const int w    = tid >> 5;
    const int lane = tid & 31;
    const int hl   = lane >> 4;      // half of wave (0/1)
    const int l16  = lane & 15;
    const int bm   = blockIdx.x * 128;
    const int bn   = blockIdx.y * 64;
    const int waveM = (w >> 1) * 32;
    const int waveN = (w & 1) * 32;

    f32x8 acc[2][2];
#pragma unroll
    for (int i = 0; i < 2; ++i)
#pragma unroll
        for (int j = 0; j < 2; ++j)
            acc[i][j] = f32x8{0.f, 0.f, 0.f, 0.f, 0.f, 0.f, 0.f, 0.f};

    for (int k0 = 0; k0 < K; k0 += 32) {
        // ---- stage A tile 128x32 (bf16, 8-elem chunks) ----
#pragma unroll
        for (int i = 0; i < 2; ++i) {
            int c  = tid + i * 256;          // 512 chunks
            int m  = c >> 2;
            int kc = (c & 3) * 8;
            unsigned short* dst = Alds + m * 40 + kc;
            if (bm + m < M) {
                const unsigned short* ap = A + (size_t)(bm + m) * lda + k0 + kc;
#if USE_ASYNC_LDS
                async_copy_b128(ap, dst);
#else
                *(u32x4*)dst = *(const u32x4*)ap;
#endif
                if (k0 + 32 < K) __builtin_prefetch((const void*)(ap + 32), 0, 1);
            } else {
                u32x4 z = {0u, 0u, 0u, 0u};
                *(u32x4*)dst = z;
            }
        }
        // ---- stage B tile 32x64, converted f32->bf16, transposed to [n][k] ----
        if (!TRANSB) {
#pragma unroll
            for (int i = 0; i < 2; ++i) {
                int c  = tid + i * 256;      // 512 chunks of 4 floats
                int k  = c >> 4;
                int n4 = (c & 15) * 4;
                f32x4 v = *(const f32x4*)(B + (size_t)(k0 + k) * ldb + bn + n4);
                Blds[(n4 + 0) * 40 + k] = f2bf(v.x);
                Blds[(n4 + 1) * 40 + k] = f2bf(v.y);
                Blds[(n4 + 2) * 40 + k] = f2bf(v.z);
                Blds[(n4 + 3) * 40 + k] = f2bf(v.w);
            }
        } else {
#pragma unroll
            for (int i = 0; i < 2; ++i) {
                int c  = tid + i * 256;
                int n  = c >> 3;
                int kc = (c & 7) * 4;
                f32x4 v = *(const f32x4*)(B + (size_t)(bn + n) * ldb + k0 + kc);
                Blds[n * 40 + kc + 0] = f2bf(v.x);
                Blds[n * 40 + kc + 1] = f2bf(v.y);
                Blds[n * 40 + kc + 2] = f2bf(v.z);
                Blds[n * 40 + kc + 3] = f2bf(v.w);
            }
        }
        async_wait0();
        __syncthreads();

        // ---- fragments (documented CDNA5 16-bit A 16x32 / B 32x16 layouts) ----
        Frag af[2], bf[2];
#pragma unroll
        for (int mt = 0; mt < 2; ++mt) {
            const unsigned short* ap = Alds + (waveM + mt * 16 + l16) * 40 + hl * 8;
            af[mt].u[0] = *(const u32x4*)ap;
            af[mt].u[1] = *(const u32x4*)(ap + 16);
        }
#pragma unroll
        for (int nt = 0; nt < 2; ++nt) {
            const unsigned short* bp = Blds + (waveN + nt * 16 + l16) * 40 + hl * 8;
            bf[nt].u[0] = *(const u32x4*)bp;
            bf[nt].u[1] = *(const u32x4*)(bp + 16);
        }
#pragma unroll
        for (int mt = 0; mt < 2; ++mt)
#pragma unroll
            for (int nt = 0; nt < 2; ++nt)
                acc[mt][nt] = wmma_bf16(af[mt].f, bf[nt].f, acc[mt][nt]);
        __syncthreads();
    }

    // ---- epilogue (C layout: lane col = l16, rows r + hl*8) ----
#pragma unroll
    for (int mt = 0; mt < 2; ++mt) {
#pragma unroll
        for (int nt = 0; nt < 2; ++nt) {
            const int col = bn + waveN + nt * 16 + l16;
            float bv = 0.f, sc = 1.f, sh = 0.f;
            if (EPI == EPI_BIAS_RES || EPI == EPI_GELU || EPI == EPI_PATCH) bv = bias[col];
            if (EPI == EPI_PATCH || EPI == EPI_BN_RELU || EPI == EPI_BN_OUT) {
                float g = bng[col], b2 = bnb[col], mm = bnm[col], vv = bnv[col];
                sc = g * rsqrtf(vv + 1e-5f);
                sh = b2 - mm * sc;
            }
#pragma unroll
            for (int r = 0; r < 8; ++r) {
                int row = bm + waveM + mt * 16 + r + hl * 8;
                if (row >= M) continue;
                float v = acc[mt][nt][r];
                if (EPI == EPI_BF16) {
                    Cb[(size_t)row * ldc + col] = f2bf(v);
                } else if (EPI == EPI_BIAS_RES) {
                    Cf[(size_t)row * ldc + col] += v + bv;
                } else if (EPI == EPI_GELU) {
                    v += bv;
                    v = 0.5f * v * (1.f + erff(v * 0.70710678118654752f));
                    Cb[(size_t)row * ldc + col] = f2bf(v);
                } else if (EPI == EPI_PATCH) {
                    v += bv;
                    v = fmaxf(v, 0.f);
                    v = v * sc + sh;
                    int bb = row / 196, ll = row % 196;
                    v += pos[(size_t)(1 + ll) * DMODEL + col];
                    Cf[((size_t)bb * NTOK + 1 + ll) * DMODEL + col] = v;
                } else if (EPI == EPI_BN_RELU) {
                    v = fmaxf(v * sc + sh, 0.f);
                    Cb[(size_t)row * ldc + col] = f2bf(v);
                } else { // EPI_BN_OUT
                    Cf[(size_t)row * ldc + col] = v * sc + sh;
                }
            }
        }
    }
}

// ---------------------------------------------------------------------------
// Fused attention: one block per (b,h); 13 waves; K,V^T,P in dynamic LDS.
// qkv: bf16 [12608, 2304]  (q|k|v each [.,h*64+d]);  out: bf16 [12608, 768]
// ---------------------------------------------------------------------------
#define KSTR 72    // Klds row stride  [208][72]
#define VSTR 232   // Vlds row stride  [64][232]   (transposed: [dv][nk])
#define PSTR 232   // Plds row stride  per-wave [16][232]

__launch_bounds__(416, 1)
__global__ void attn_kernel(const unsigned short* __restrict__ qkv,
                            unsigned short* __restrict__ obf) {
    extern __shared__ __align__(16) unsigned short smem[];
    unsigned short* Klds = smem;                       // 208*72  = 14976
    unsigned short* Vlds = Klds + 208 * KSTR;          // 64*232  = 14848
    unsigned short* Plds = Vlds + 64 * VSTR;           // 13*16*232

    const int bh = blockIdx.x;
    const int b  = bh / NHEAD;
    const int h  = bh % NHEAD;
    const int tid  = threadIdx.x;
    const int w    = tid >> 5;
    const int lane = tid & 31;
    const int hl   = lane >> 4;
    const int l16  = lane & 15;
    const size_t base = (size_t)b * NTOK * 2304 + h * HD;

    // ---- K -> LDS [nk][d], zero-padded to 208 rows (async path when available) ----
    for (int idx = tid; idx < 208 * 8; idx += 416) {
        int nk = idx >> 3, dc = (idx & 7) * 8;
        unsigned short* dst = Klds + nk * KSTR + dc;
        if (nk < NTOK) {
#if USE_ASYNC_LDS
            async_copy_b128(qkv + base + DMODEL + (size_t)nk * 2304 + dc, dst);
#else
            *(u32x4*)dst = *(const u32x4*)(qkv + base + DMODEL + (size_t)nk * 2304 + dc);
#endif
        } else {
            u32x4 z = {0u, 0u, 0u, 0u};
            *(u32x4*)dst = z;
        }
    }
    // ---- V^T -> LDS [dv][nk], nk padded to 224 (transpose: element-wise) ----
    for (int idx = tid; idx < 224 * 64; idx += 416) {
        int nk = idx >> 6, dv = idx & 63;
        unsigned short v = 0;
        if (nk < NTOK) v = qkv[base + 2 * DMODEL + (size_t)nk * 2304 + dv];
        Vlds[dv * VSTR + nk] = v;
    }
    // ---- zero P pad columns 208..223 for this wave ----
    unsigned short* Pw = Plds + w * 16 * PSTR;
    for (int i = lane; i < 16 * 16; i += 32)
        Pw[(i >> 4) * PSTR + 208 + (i & 15)] = 0;
    async_wait0();
    __syncthreads();

    // ---- Q fragments straight from global (reused across all 13 K tiles) ----
    const int qrow = w * 16 + l16;         // 0..207 local query row
    const bool qvalid = qrow < NTOK;
    Frag qa[2];
#pragma unroll
    for (int ks = 0; ks < 2; ++ks) {
        u32x4 z = {0u, 0u, 0u, 0u};
        if (qvalid) {
            const unsigned short* qp = qkv + base + (size_t)qrow * 2304 + ks * 32 + hl * 8;
            qa[ks].u[0] = *(const u32x4*)qp;
            qa[ks].u[1] = *(const u32x4*)(qp + 16);
        } else { qa[ks].u[0] = z; qa[ks].u[1] = z; }
    }

    // ---- S = Q * K^T : 13 col-tiles x 2 k-steps ----
    f32x8 sacc[13];
#pragma unroll
    for (int j = 0; j < 13; ++j) {
        f32x8 c = f32x8{0.f, 0.f, 0.f, 0.f, 0.f, 0.f, 0.f, 0.f};
#pragma unroll
        for (int ks = 0; ks < 2; ++ks) {
            Frag kb;
            const unsigned short* kp = Klds + (j * 16 + l16) * KSTR + ks * 32 + hl * 8;
            kb.u[0] = *(const u32x4*)kp;
            kb.u[1] = *(const u32x4*)(kp + 16);
            c = wmma_bf16(qa[ks].f, kb.f, c);
        }
        sacc[j] = c;
    }

    // ---- softmax over rows (16-lane xor reductions within wave32 halves) ----
#pragma unroll
    for (int r = 0; r < 8; ++r) {
        float vals[13];
        float m = -1e30f;
#pragma unroll
        for (int j = 0; j < 13; ++j) {
            int col = j * 16 + l16;
            float v = sacc[j][r] * 0.125f;   // SCALE = 64^-0.5
            vals[j] = v;
            if (col < NTOK) m = fmaxf(m, v);
        }
        m = fmaxf(m, __shfl_xor(m, 1, 32));
        m = fmaxf(m, __shfl_xor(m, 2, 32));
        m = fmaxf(m, __shfl_xor(m, 4, 32));
        m = fmaxf(m, __shfl_xor(m, 8, 32));
        float s = 0.f;
#pragma unroll
        for (int j = 0; j < 13; ++j) {
            int col = j * 16 + l16;
            float p = (col < NTOK) ? __expf(vals[j] - m) : 0.f;
            vals[j] = p;
            s += p;
        }
        s += __shfl_xor(s, 1, 32);
        s += __shfl_xor(s, 2, 32);
        s += __shfl_xor(s, 4, 32);
        s += __shfl_xor(s, 8, 32);
        float inv = 1.f / s;
        int prow = r + hl * 8;
#pragma unroll
        for (int j = 0; j < 13; ++j)
            Pw[prow * PSTR + j * 16 + l16] = f2bf(vals[j] * inv);
    }

    // ---- O = P * V : 7 k-steps (k padded to 224) x 4 col-tiles ----
    f32x8 oacc[4];
#pragma unroll
    for (int nt = 0; nt < 4; ++nt)
        oacc[nt] = f32x8{0.f, 0.f, 0.f, 0.f, 0.f, 0.f, 0.f, 0.f};
#pragma unroll
    for (int ks = 0; ks < 7; ++ks) {
        Frag pa;
        const unsigned short* pp = Pw + l16 * PSTR + ks * 32 + hl * 8;
        pa.u[0] = *(const u32x4*)pp;
        pa.u[1] = *(const u32x4*)(pp + 16);
#pragma unroll
        for (int nt = 0; nt < 4; ++nt) {
            Frag vb;
            const unsigned short* vp = Vlds + (nt * 16 + l16) * VSTR + ks * 32 + hl * 8;
            vb.u[0] = *(const u32x4*)vp;
            vb.u[1] = *(const u32x4*)(vp + 16);
            oacc[nt] = wmma_bf16(pa.f, vb.f, oacc[nt]);
        }
    }

    // ---- store O ----
#pragma unroll
    for (int nt = 0; nt < 4; ++nt) {
#pragma unroll
        for (int r = 0; r < 8; ++r) {
            int qr = w * 16 + r + hl * 8;
            if (qr < NTOK)
                obf[((size_t)b * NTOK + qr) * DMODEL + h * HD + nt * 16 + l16] = f2bf(oacc[nt][r]);
        }
    }
}

// ---------------------------------------------------------------------------
// LayerNorm: one block (256 thr) per row of 768; f32 in -> bf16 out.
// row_in = blockIdx.x * rowmul (rowmul=197 extracts cls tokens), out row = blockIdx.x
// ---------------------------------------------------------------------------
__launch_bounds__(256)
__global__ void ln_kernel(const float* __restrict__ x, unsigned short* __restrict__ y,
                          const float* __restrict__ g, const float* __restrict__ b,
                          int rowmul) {
    const size_t row = (size_t)blockIdx.x * rowmul;
    const float* xr = x + row * DMODEL;
    const int tid = threadIdx.x;
    float v0 = xr[tid], v1 = xr[tid + 256], v2 = xr[tid + 512];
    float s = v0 + v1 + v2;
    float q = v0 * v0 + v1 * v1 + v2 * v2;
    __shared__ float rs[8], rq[8];
    for (int off = 16; off >= 1; off >>= 1) {
        s += __shfl_xor(s, off, 32);
        q += __shfl_xor(q, off, 32);
    }
    int w = tid >> 5, lane = tid & 31;
    if (lane == 0) { rs[w] = s; rq[w] = q; }
    __syncthreads();
    if (w == 0) {
        float ss = (lane < 8) ? rs[lane] : 0.f;
        float qq = (lane < 8) ? rq[lane] : 0.f;
        for (int off = 4; off >= 1; off >>= 1) {
            ss += __shfl_xor(ss, off, 32);
            qq += __shfl_xor(qq, off, 32);
        }
        if (lane == 0) { rs[0] = ss; rq[0] = qq; }
    }
    __syncthreads();
    float mean = rs[0] * (1.f / 768.f);
    float var  = rq[0] * (1.f / 768.f) - mean * mean;
    float inv  = rsqrtf(var + 1e-6f);
    unsigned short* yr = y + (size_t)blockIdx.x * DMODEL;
    yr[tid]       = f2bf((v0 - mean) * inv * g[tid]       + b[tid]);
    yr[tid + 256] = f2bf((v1 - mean) * inv * g[tid + 256] + b[tid + 256]);
    yr[tid + 512] = f2bf((v2 - mean) * inv * g[tid + 512] + b[tid + 512]);
}

// ---------------------------------------------------------------------------
// PatchEmbed conv1 (k=5,s=5) + ReLU + BN -> bf16 [64][256][200]
// ---------------------------------------------------------------------------
__launch_bounds__(256)
__global__ void conv1_kernel(const float* __restrict__ x, const float* __restrict__ wgt,
                             const float* __restrict__ bias,
                             const float* __restrict__ g, const float* __restrict__ bb,
                             const float* __restrict__ m, const float* __restrict__ v,
                             unsigned short* __restrict__ out) {
    int idx = blockIdx.x * 256 + threadIdx.x;
    if (idx >= 64 * 256 * 200) return;
    int l  = idx % 200;
    int co = (idx / 200) % 256;
    int b  = idx / (200 * 256);
    float acc = bias[co];
    const float* xb = x + (size_t)b * 12000 + l * 5;
    const float* wc = wgt + co * 60;
#pragma unroll
    for (int ci = 0; ci < 12; ++ci)
#pragma unroll
        for (int t = 0; t < 5; ++t)
            acc += xb[ci * 1000 + t] * wc[ci * 5 + t];
    acc = fmaxf(acc, 0.f);
    float sc = g[co] * rsqrtf(v[co] + 1e-5f);
    acc = (acc - m[co]) * sc + bb[co];
    out[idx] = f2bf(acc);
}

// im2col for conv2 (k=5,s=1): A2[b*196+l][ci*5+t] = h1[b][ci][l+t]
__launch_bounds__(256)
__global__ void im2col_kernel(const unsigned short* __restrict__ h1,
                              unsigned short* __restrict__ a) {
    size_t idx = (size_t)blockIdx.x * 256 + threadIdx.x;
    if (idx >= (size_t)12544 * 1280) return;
    int kk = (int)(idx % 1280);
    int rr = (int)(idx / 1280);
    int l  = rr % 196;
    int b  = rr / 196;
    int ci = kk / 5, t = kk % 5;
    a[idx] = h1[(size_t)b * 256 * 200 + ci * 200 + l + t];
}

// cls row init: tok[b*197 + 0, :] = cls_token + pos_embed[0]
__launch_bounds__(256)
__global__ void cls_init_kernel(const float* __restrict__ cls_tok,
                                const float* __restrict__ pos,
                                float* __restrict__ tok) {
    int c = blockIdx.x * 256 + threadIdx.x;
    if (c >= DMODEL) return;
    float v = cls_tok[c] + pos[c];
    for (int b = 0; b < NBATCH; ++b)
        tok[(size_t)b * NTOK * DMODEL + c] = v;
}

// ---------------------------------------------------------------------------
// Host-side orchestration
// ---------------------------------------------------------------------------
extern "C" void kernel_launch(void* const* d_in, const int* in_sizes, int n_in,
                              void* d_out, int out_size, void* d_ws, size_t ws_size,
                              hipStream_t stream) {
    (void)in_sizes; (void)n_in; (void)out_size; (void)ws_size;
    const float* x       = (const float*)d_in[0];
    const float* conv1_w = (const float*)d_in[1];
    const float* conv1_b = (const float*)d_in[2];
    const float* bn1_g   = (const float*)d_in[3];
    const float* bn1_b   = (const float*)d_in[4];
    const float* bn1_m   = (const float*)d_in[5];
    const float* bn1_v   = (const float*)d_in[6];
    const float* conv2_w = (const float*)d_in[7];
    const float* conv2_b = (const float*)d_in[8];
    const float* bn2_g   = (const float*)d_in[9];
    const float* bn2_b   = (const float*)d_in[10];
    const float* bn2_m   = (const float*)d_in[11];
    const float* bn2_v   = (const float*)d_in[12];
    const float* cls_tok = (const float*)d_in[13];
    const float* pos_emb = (const float*)d_in[14];
    const float* ln1_g   = (const float*)d_in[15];
    const float* ln1_b   = (const float*)d_in[16];
    const float* qkv_w   = (const float*)d_in[17];
    const float* proj_w  = (const float*)d_in[18];
    const float* proj_b  = (const float*)d_in[19];
    const float* ln2_g   = (const float*)d_in[20];
    const float* ln2_b   = (const float*)d_in[21];
    const float* fc1_w   = (const float*)d_in[22];
    const float* fc1_b   = (const float*)d_in[23];
    const float* fc2_w   = (const float*)d_in[24];
    const float* fc2_b   = (const float*)d_in[25];
    const float* norm_g  = (const float*)d_in[26];
    const float* norm_b  = (const float*)d_in[27];
    const float* p1_w    = (const float*)d_in[28];
    const float* p1bn_g  = (const float*)d_in[29];
    const float* p1bn_b  = (const float*)d_in[30];
    const float* p1bn_m  = (const float*)d_in[31];
    const float* p1bn_v  = (const float*)d_in[32];
    const float* p2_w    = (const float*)d_in[33];
    const float* p2bn_g  = (const float*)d_in[34];
    const float* p2bn_b  = (const float*)d_in[35];
    const float* p2bn_m  = (const float*)d_in[36];
    const float* p2bn_v  = (const float*)d_in[37];

    // workspace carve-out (lifetimes: big = im2col -> obf -> hbf, per phase)
    char* wsb = (char*)d_ws;
    size_t off = 0;
    auto take = [&](size_t bytes) -> char* {
        char* p = wsb + off;
        off = (off + bytes + 255) & ~(size_t)255;
        return p;
    };
    float*          tok   = (float*)         take((size_t)MROWS * DMODEL * 4);
    unsigned short* ybf   = (unsigned short*)take((size_t)MROWS * DMODEL * 2);
    unsigned short* qkvbf = (unsigned short*)take((size_t)MROWS * 2304 * 2);
    unsigned short* big   = (unsigned short*)take((size_t)MROWS * 3072 * 2); // im2col/obf/hbf
    unsigned short* h1bf  = (unsigned short*)take((size_t)64 * 256 * 200 * 2);
    unsigned short* clsbf = (unsigned short*)take((size_t)64 * DMODEL * 2);
    unsigned short* z1bf  = (unsigned short*)take((size_t)64 * DMODEL * 2);

    const int attn_lds = (208 * KSTR + 64 * VSTR + 13 * 16 * PSTR) * 2; // 156160 B
    (void)hipFuncSetAttribute(reinterpret_cast<const void*>(attn_kernel),
                              hipFuncAttributeMaxDynamicSharedMemorySize, attn_lds);

    // ---- patch embed ----
    conv1_kernel<<<12800, 256, 0, stream>>>(x, conv1_w, conv1_b, bn1_g, bn1_b, bn1_m, bn1_v, h1bf);
    im2col_kernel<<<62720, 256, 0, stream>>>(h1bf, big);
    cls_init_kernel<<<3, 256, 0, stream>>>(cls_tok, pos_emb, tok);
    // conv2 as GEMM (weights [O][I*5] => TRANSB), fused ReLU+BN+pos scatter into tok
    gemm_kernel<EPI_PATCH, true><<<dim3(98, 12), 256, 0, stream>>>(
        big, conv2_w, tok, nullptr, conv2_b, bn2_g, bn2_b, bn2_m, bn2_v, pos_emb,
        12544, 768, 1280, 1280, 1280, 768);

    // ---- transformer layers ----
    for (int i = 0; i < 12; ++i) {
        const float* qw  = qkv_w  + (size_t)i * DMODEL * 2304;
        const float* pw  = proj_w + (size_t)i * DMODEL * DMODEL;
        const float* pb  = proj_b + (size_t)i * DMODEL;
        const float* f1w = fc1_w  + (size_t)i * DMODEL * 3072;
        const float* f1b = fc1_b  + (size_t)i * 3072;
        const float* f2w = fc2_w  + (size_t)i * 3072 * DMODEL;
        const float* f2b = fc2_b  + (size_t)i * DMODEL;

        ln_kernel<<<MROWS, 256, 0, stream>>>(tok, ybf, ln1_g + i * DMODEL, ln1_b + i * DMODEL, 1);
        gemm_kernel<EPI_BF16, false><<<dim3(99, 36), 256, 0, stream>>>(
            ybf, qw, nullptr, qkvbf, nullptr, nullptr, nullptr, nullptr, nullptr, nullptr,
            MROWS, 2304, 768, 768, 2304, 2304);
        attn_kernel<<<NBATCH * NHEAD, 416, attn_lds, stream>>>(qkvbf, big);
        gemm_kernel<EPI_BIAS_RES, false><<<dim3(99, 12), 256, 0, stream>>>(
            big, pw, tok, nullptr, pb, nullptr, nullptr, nullptr, nullptr, nullptr,
            MROWS, 768, 768, 768, 768, 768);
        ln_kernel<<<MROWS, 256, 0, stream>>>(tok, ybf, ln2_g + i * DMODEL, ln2_b + i * DMODEL, 1);
        gemm_kernel<EPI_GELU, false><<<dim3(99, 48), 256, 0, stream>>>(
            ybf, f1w, nullptr, big, f1b, nullptr, nullptr, nullptr, nullptr, nullptr,
            MROWS, 3072, 768, 768, 3072, 3072);
        gemm_kernel<EPI_BIAS_RES, false><<<dim3(99, 12), 256, 0, stream>>>(
            big, f2w, tok, nullptr, f2b, nullptr, nullptr, nullptr, nullptr, nullptr,
            MROWS, 768, 3072, 3072, 768, 768);
    }

    // ---- head: final LN on cls tokens, p1 (BN+ReLU), p2 (BN) ----
    ln_kernel<<<64, 256, 0, stream>>>(tok, clsbf, norm_g, norm_b, NTOK);
    gemm_kernel<EPI_BN_RELU, false><<<dim3(1, 12), 256, 0, stream>>>(
        clsbf, p1_w, nullptr, z1bf, nullptr, p1bn_g, p1bn_b, p1bn_m, p1bn_v, nullptr,
        64, 768, 768, 768, 768, 768);
    gemm_kernel<EPI_BN_OUT, false><<<dim3(1, 8), 256, 0, stream>>>(
        z1bf, p2_w, (float*)d_out, nullptr, nullptr, p2bn_g, p2bn_b, p2bn_m, p2bn_v, nullptr,
        64, 512, 768, 768, 512, 512);
}
